// WindowAttention_48722109005873
// MI455X (gfx1250) — compile-verified, matrix operand order checked by reading, not run on hardware
//
#include <hip/hip_runtime.h>

#define BN 4
#define LL 2048
#define CC 512
#define HH 8
#define DD 64

typedef __bf16 bf16;
typedef __attribute__((ext_vector_type(8)))  __bf16 v8bf;
typedef __attribute__((ext_vector_type(16))) __bf16 v16bf;
typedef __attribute__((ext_vector_type(8)))  float  v8f;
typedef __attribute__((ext_vector_type(4)))  float  v4f;

// Load a 32-wide-K bf16 WMMA fragment from an LDS tile.
// ISA layout (16-bit A 16x32 / B 32x16): lane half g = lane/16 holds
// K = [8g .. 8g+7] in elems 0..7 and K = [16+8g .. 16+8g+7] in elems 8..15.
__device__ __forceinline__ v16bf frag_ld(const bf16* p, int row, int stride, int g) {
  const v8bf lo = *(const v8bf*)(p + row * stride + 8 * g);
  const v8bf hi = *(const v8bf*)(p + row * stride + 16 + 8 * g);
  v16bf r;
#pragma unroll
  for (int i = 0; i < 8; ++i) { r[i] = lo[i]; r[i + 8] = hi[i]; }
  return r;
}

__device__ __forceinline__ v8f wmma_bf16(v16bf a, v16bf b, v8f c) {
  return __builtin_amdgcn_wmma_f32_16x16x32_bf16(false, a, false, b, (short)0, c,
                                                 false, false);
}

// CDNA5 async copy: global -> LDS, 128b, tracked by ASYNCcnt (no VGPR staging).
__device__ __forceinline__ void async_b128(const bf16* gsrc, bf16* ldst) {
  unsigned l = (unsigned)(size_t)(void*)ldst;           // low 32b = LDS offset
  unsigned long long ga = (unsigned long long)(size_t)gsrc;
  asm volatile("global_load_async_to_lds_b128 %0, %1, off"
               :: "v"(l), "v"(ga) : "memory");
}

__device__ __forceinline__ void wait_async0() {
#if __has_builtin(__builtin_amdgcn_s_wait_asynccnt)
  __builtin_amdgcn_s_wait_asynccnt(0);
#else
  asm volatile("s_wait_asynccnt 0x0" ::: "memory");
#endif
}

// ---------------------------------------------------------------------------
// Kernel 1: qkv = x @ qkv_w^T + qkv_b, split heads, RoPE on q/k.
// Q,K stored bf16 (B,H,L,D); V stored TRANSPOSED bf16 (B,H,D,L) so the
// attention kernel's V tiles are pure row-major b128 async copies.
// Block tile: 128(M) x 64(N = one (t,h) slice); wave owns 32 rows.
// ---------------------------------------------------------------------------
__global__ __launch_bounds__(128)
void qkv_rope_kernel(const float* __restrict__ x, const float* __restrict__ qkv_w,
                     const float* __restrict__ qkv_b, const float* __restrict__ cosT,
                     const float* __restrict__ sinT, bf16* __restrict__ Qr,
                     bf16* __restrict__ Kr, bf16* __restrict__ VtW) {
  __shared__ alignas(16) bf16 As[128 * 32];
  __shared__ alignas(16) bf16 Bs[64 * 32];
  const int tid = threadIdx.x;
  const int wave = tid >> 5, lane = tid & 31;
  const int ln = lane & 15, g = lane >> 4;
  const int m0 = blockIdx.x * 128;
  const int nb = blockIdx.y;          // 0..23
  const int n0 = nb * 64;
  const int t  = nb >> 3;             // 0=q 1=k 2=v
  const int h  = nb & 7;

  v8f acc[2][4] = {};

  for (int k0 = 0; k0 < CC; k0 += 32) {
    __syncthreads();
#pragma unroll
    for (int i = 0; i < 8; ++i) {     // x tile: 128x32 f32 -> bf16
      int e = i * 512 + tid * 4;
      int row = e >> 5, col = e & 31;
      v4f v = *(const v4f*)(x + (size_t)(m0 + row) * CC + k0 + col);
      bf16* d = As + row * 32 + col;
      d[0] = (bf16)v[0]; d[1] = (bf16)v[1]; d[2] = (bf16)v[2]; d[3] = (bf16)v[3];
    }
#pragma unroll
    for (int i = 0; i < 4; ++i) {     // qkv_w tile: 64x32 f32 -> bf16
      int e = i * 512 + tid * 4;
      int row = e >> 5, col = e & 31;
      v4f v = *(const v4f*)(qkv_w + (size_t)(n0 + row) * CC + k0 + col);
      bf16* d = Bs + row * 32 + col;
      d[0] = (bf16)v[0]; d[1] = (bf16)v[1]; d[2] = (bf16)v[2]; d[3] = (bf16)v[3];
    }
    __syncthreads();
    v16bf a0 = frag_ld(As, 32 * wave + ln, 32, g);
    v16bf a1 = frag_ld(As, 32 * wave + 16 + ln, 32, g);
    v16bf b[4];
#pragma unroll
    for (int j = 0; j < 4; ++j) b[j] = frag_ld(Bs, 16 * j + ln, 32, g);
#pragma unroll
    for (int j = 0; j < 4; ++j) {
      acc[0][j] = wmma_bf16(a0, b[j], acc[0][j]);
      acc[1][j] = wmma_bf16(a1, b[j], acc[1][j]);
    }
  }

  // Epilogue: bias + RoPE. Cols d and d+32 live in acc[rb][j] / acc[rb][j+2]
  // of the same lane (C-layout col = 16j + lane%16): RoPE mixes in registers.
#pragma unroll
  for (int rb = 0; rb < 2; ++rb) {
#pragma unroll
    for (int j = 0; j < 2; ++j) {
      const int d = 16 * j + ln;      // 0..31
      const float bia = qkv_b[n0 + d];
      const float bib = qkv_b[n0 + d + 32];
#pragma unroll
      for (int r = 0; r < 8; ++r) {
        const int mrow = m0 + 32 * wave + 16 * rb + 8 * g + r;
        const int lseq = mrow & (LL - 1);
        const int bb   = mrow >> 11;
        float av = acc[rb][j][r] + bia;
        float bv = acc[rb][j + 2][r] + bib;
        if (t < 2) {
          const float cv = cosT[lseq * DD + d];   // cos[d] == cos[d+32]
          const float sv = sinT[lseq * DD + d];
          float o1 = av * cv - bv * sv;
          float o2 = bv * cv + av * sv;
          bf16* dst = (t == 0) ? Qr : Kr;
          const size_t off = (((size_t)bb * HH + h) * LL + lseq) * DD + d;
          dst[off]      = (bf16)o1;
          dst[off + 32] = (bf16)o2;
        } else {                                  // V: transposed (B,H,D,L)
          const size_t off = (((size_t)bb * HH + h) * DD + d) * LL + lseq;
          VtW[off]           = (bf16)av;
          VtW[off + 32 * LL] = (bf16)bv;
        }
      }
    }
  }
}

// ---------------------------------------------------------------------------
// Kernel 2: flash attention. 64-query tile/block; 64-key tiles streamed with
// CDNA5 async global->LDS copies, double-buffered (compute tile i overlaps
// the ASYNCcnt-tracked DMA of tile i+1). Wave owns 16 query rows.
// ---------------------------------------------------------------------------
__global__ __launch_bounds__(128)
void attn_kernel(const bf16* __restrict__ Qr, const bf16* __restrict__ Kr,
                 const bf16* __restrict__ VtW, bf16* __restrict__ Ao) {
  __shared__ alignas(16) bf16 Qs[64 * 64];
  __shared__ alignas(16) bf16 Ks[2][64 * 64];    // row-major keys x d
  __shared__ alignas(16) bf16 Vts[2][64 * 64];   // row-major d x keys
  __shared__ alignas(16) bf16 Ps[4][16 * 64];    // per-wave P staging

  const int tid = threadIdx.x;
  const int wave = tid >> 5, lane = tid & 31;
  const int ln = lane & 15, g = lane >> 4;
  const int q0 = blockIdx.x * 64;
  const int bh = blockIdx.y;
  const bf16* Qbase  = Qr  + (size_t)bh * LL * DD;
  const bf16* Kbase  = Kr  + (size_t)bh * LL * DD;
  const bf16* VtBase = VtW + (size_t)bh * DD * LL;

  auto issue_tile = [&](int ti, int bufi) {
#pragma unroll
    for (int i = 0; i < 4; ++i) {                // K tile: direct copy
      int j = tid + 128 * i;                     // b128 index, 512 total
      async_b128(Kbase + (size_t)ti * 64 * DD + j * 8, &Ks[bufi][j * 8]);
    }
#pragma unroll
    for (int i = 0; i < 4; ++i) {                // V^T tile: rows d, cols key
      int j = tid + 128 * i;
      int d = j >> 3, kk = (j & 7) * 8;
      async_b128(VtBase + (size_t)d * LL + ti * 64 + kk, &Vts[bufi][d * 64 + kk]);
    }
  };

  // prologue: Q tile + key-tile 0 via async DMA
#pragma unroll
  for (int i = 0; i < 4; ++i) {
    int j = tid + 128 * i;
    async_b128(Qbase + (size_t)q0 * DD + j * 8, Qs + j * 8);
  }
  issue_tile(0, 0);
  wait_async0();
  __syncthreads();

  v16bf qa[2];
  qa[0] = frag_ld(Qs,      16 * wave + ln, 64, g);   // d 0..31
  qa[1] = frag_ld(Qs + 32, 16 * wave + ln, 64, g);   // d 32..63

  v8f o[4] = {};
  float mrow[8], lrow[8];
#pragma unroll
  for (int r = 0; r < 8; ++r) { mrow[r] = -1e30f; lrow[r] = 0.f; }

  const int T = LL / 64;                             // 32 key tiles
  for (int it = 0; it < T; ++it) {
    const int cur = it & 1;
    if (it + 1 < T) issue_tile(it + 1, cur ^ 1);     // overlap next tile's DMA

    // S = (Q K^T) * scale  -- batch fragment loads so one wait covers 4 WMMAs
    v8f s[4] = {};
#pragma unroll
    for (int ks = 0; ks < 2; ++ks) {
      v16bf kb[4];
#pragma unroll
      for (int jn = 0; jn < 4; ++jn)
        kb[jn] = frag_ld(&Ks[cur][32 * ks], 16 * jn + ln, 64, g);
#pragma unroll
      for (int jn = 0; jn < 4; ++jn) s[jn] = wmma_bf16(qa[ks], kb[jn], s[jn]);
    }

    // online softmax: rows live in 16-lane halves -> xor-shuffle reduce
#pragma unroll
    for (int r = 0; r < 8; ++r) {
      float tmax = -1e30f;
#pragma unroll
      for (int jn = 0; jn < 4; ++jn) {
        s[jn][r] *= 0.125f;                          // 1/sqrt(D)
        tmax = fmaxf(tmax, s[jn][r]);
      }
#pragma unroll
      for (int off = 8; off >= 1; off >>= 1)
        tmax = fmaxf(tmax, __shfl_xor(tmax, off, 32));
      float mnew = fmaxf(mrow[r], tmax);
      float corr = __expf(mrow[r] - mnew);
      float psum = 0.f;
#pragma unroll
      for (int jn = 0; jn < 4; ++jn) {
        float p = __expf(s[jn][r] - mnew);
        s[jn][r] = p;
        psum += p;
      }
#pragma unroll
      for (int off = 8; off >= 1; off >>= 1)
        psum += __shfl_xor(psum, off, 32);
      lrow[r] = lrow[r] * corr + psum;
      mrow[r] = mnew;
#pragma unroll
      for (int jn = 0; jn < 4; ++jn) o[jn][r] *= corr;
    }

    // C-layout -> A-fragment layout via per-wave LDS (same-wave DS in-order)
#pragma unroll
    for (int jn = 0; jn < 4; ++jn)
#pragma unroll
      for (int r = 0; r < 8; ++r)
        Ps[wave][(8 * g + r) * 64 + 16 * jn + ln] = (bf16)s[jn][r];

    // O += P @ V
#pragma unroll
    for (int ks = 0; ks < 2; ++ks) {
      v16bf pa = frag_ld(&Ps[wave][32 * ks], ln, 64, g);
      v16bf vb[4];
#pragma unroll
      for (int jd = 0; jd < 4; ++jd)
        vb[jd] = frag_ld(&Vts[cur][32 * ks], 16 * jd + ln, 64, g);
#pragma unroll
      for (int jd = 0; jd < 4; ++jd) o[jd] = wmma_bf16(pa, vb[jd], o[jd]);
    }

    wait_async0();        // next tile's DMA has landed
    __syncthreads();      // all waves done with buf[cur] before it's reused
  }

  // finalize; store (B, L, H*D) so projection reads row-major
  const int bb = bh >> 3, h = bh & 7;
#pragma unroll
  for (int jd = 0; jd < 4; ++jd)
#pragma unroll
    for (int r = 0; r < 8; ++r) {
      int row = 16 * wave + 8 * g + r;
      float val = o[jd][r] / lrow[r];
      size_t off = ((size_t)bb * LL + q0 + row) * CC + h * DD + 16 * jd + ln;
      Ao[off] = (bf16)val;
    }
}

// ---------------------------------------------------------------------------
// Kernel 3: out = Ao @ proj_w^T + proj_b (fp32 output). 128x64 block tile.
// ---------------------------------------------------------------------------
__global__ __launch_bounds__(128)
void proj_kernel(const bf16* __restrict__ Ao, const float* __restrict__ proj_w,
                 const float* __restrict__ proj_b, float* __restrict__ out) {
  __shared__ alignas(16) bf16 As[128 * 32];
  __shared__ alignas(16) bf16 Bs[64 * 32];
  const int tid = threadIdx.x;
  const int wave = tid >> 5, lane = tid & 31;
  const int ln = lane & 15, g = lane >> 4;
  const int m0 = blockIdx.x * 128;
  const int n0 = blockIdx.y * 64;

  v8f acc[2][4] = {};

  for (int k0 = 0; k0 < CC; k0 += 32) {
    __syncthreads();
#pragma unroll
    for (int i = 0; i < 4; ++i) {     // A (bf16): 512 b128 copies
      int j = tid + 128 * i;
      int row = j >> 2, col = (j & 3) * 8;
      *(uint4*)(As + row * 32 + col) =
          *(const uint4*)(Ao + (size_t)(m0 + row) * CC + k0 + col);
    }
#pragma unroll
    for (int i = 0; i < 4; ++i) {     // proj_w: 64x32 f32 -> bf16
      int e = i * 512 + tid * 4;
      int row = e >> 5, col = e & 31;
      v4f v = *(const v4f*)(proj_w + (size_t)(n0 + row) * CC + k0 + col);
      bf16* d = Bs + row * 32 + col;
      d[0] = (bf16)v[0]; d[1] = (bf16)v[1]; d[2] = (bf16)v[2]; d[3] = (bf16)v[3];
    }
    __syncthreads();
    v16bf a0 = frag_ld(As, 32 * wave + ln, 32, g);
    v16bf a1 = frag_ld(As, 32 * wave + 16 + ln, 32, g);
    v16bf b[4];
#pragma unroll
    for (int j = 0; j < 4; ++j) b[j] = frag_ld(Bs, 16 * j + ln, 32, g);
#pragma unroll
    for (int j = 0; j < 4; ++j) {
      acc[0][j] = wmma_bf16(a0, b[j], acc[0][j]);
      acc[1][j] = wmma_bf16(a1, b[j], acc[1][j]);
    }
  }

#pragma unroll
  for (int rb = 0; rb < 2; ++rb)
#pragma unroll
    for (int j = 0; j < 4; ++j) {
      const float bias = proj_b[n0 + 16 * j + ln];
#pragma unroll
      for (int r = 0; r < 8; ++r) {
        int mrow = m0 + 32 * wave + 16 * rb + 8 * g + r;
        out[(size_t)mrow * CC + n0 + 16 * j + ln] = acc[rb][j][r] + bias;
      }
    }
}

extern "C" void kernel_launch(void* const* d_in, const int* in_sizes, int n_in,
                              void* d_out, int out_size, void* d_ws, size_t ws_size,
                              hipStream_t stream) {
  (void)in_sizes; (void)n_in; (void)out_size; (void)ws_size;
  const float* x      = (const float*)d_in[0];
  const float* qkv_w  = (const float*)d_in[1];
  const float* qkv_b  = (const float*)d_in[2];
  const float* proj_w = (const float*)d_in[3];
  const float* proj_b = (const float*)d_in[4];
  const float* cosT   = (const float*)d_in[5];
  const float* sinT   = (const float*)d_in[6];
  float* out = (float*)d_out;

  const size_t nBHLD = (size_t)BN * HH * LL * DD;   // 4,194,304 elems
  bf16* Qr  = (bf16*)d_ws;
  bf16* Kr  = Qr + nBHLD;
  bf16* VtW = Kr + nBHLD;                           // (B,H,D,L) transposed V
  bf16* Ao  = VtW + nBHLD;                          // (B,L,C)

  qkv_rope_kernel<<<dim3(64, 24), 128, 0, stream>>>(x, qkv_w, qkv_b, cosT, sinT,
                                                    Qr, Kr, VtW);
  attn_kernel<<<dim3(32, 32), 128, 0, stream>>>(Qr, Kr, VtW, Ao);
  proj_kernel<<<dim3(64, 8), 128, 0, stream>>>(Ao, proj_w, proj_b, out);
}
